// CaserQueryEncoder_10445360464145
// MI455X (gfx1250) — compile-verified, matrix operand order checked by reading, not run on hardware
//
#include <hip/hip_runtime.h>

// ---------------------------------------------------------------------------
// CDNA5 (gfx1250, wave32) Caser query encoder.
// Dominant cost: masked horizontal conv (~46 G MACs) -> bf16 WMMA 16x16x32,
// E window staged in LDS (async-to-LDS when available), NST-specialized loops.
// ---------------------------------------------------------------------------

typedef __attribute__((ext_vector_type(16))) __bf16 v16bf;
typedef __attribute__((ext_vector_type(8)))  __bf16 v8bf;
typedef __attribute__((ext_vector_type(4)))  __bf16 v4bf;
typedef __attribute__((ext_vector_type(8)))  float  v8f;
typedef __attribute__((ext_vector_type(4)))  int    v4i;

#define CB   1024
#define CL   50
#define CD   128
#define CNV  8
#define CNH  16
#define OV_DIM 1024          // NV*D
#define OH_DIM 800           // L*NH
#define O_DIM  1824          // OV_DIM + OH_DIM
#define KC_FC  57            // O_DIM / 32
#define EROWS  128           // padded rows per batch in global Epad
#define SROWS  66            // rows staged in LDS (max access = 64)
#define SPITCH 136           // LDS row pitch in elems (68 dwords -> no bank conflicts)

#if defined(__AMDGCN__) && __has_builtin(__builtin_amdgcn_global_load_async_to_lds_b128) && __has_builtin(__builtin_amdgcn_s_wait_asynccnt)
#define HAVE_ASYNC 1
#else
#define HAVE_ASYNC 0
#endif

#if HAVE_ASYNC
// Builtin prototype (from clang diagnostic): arg0 = v4i addrspace(1)* (global,
// printed as "__device__"), arg1 = v4i addrspace(3)* (LDS), then imm offset,
// imm cpol. AS3 pointers are 32-bit; low 32 bits of a generic LDS address are
// the LDS offset, so we go through integer casts.
typedef __attribute__((address_space(1))) v4i g_v4i;
typedef __attribute__((address_space(3))) v4i l_v4i;
__device__ __forceinline__ void async_copy16(const void* g, void* l) {
  __builtin_amdgcn_global_load_async_to_lds_b128(
      (g_v4i*)(unsigned long long)g,
      (l_v4i*)(unsigned int)(unsigned long long)l, 0, 0);
}
#endif

// K-slot mapping inside a 32-wide chunk for 16-bit WMMA A/B fragments:
// lane half h (lane>>4), element e (0..15) -> K = 16*(e>>3) + 8*h + (e&7)
__device__ __forceinline__ int kslot(int e, int half) {
  return ((e >> 3) << 4) + (half << 3) + (e & 7);
}

// Load a 16-bf16 A fragment slice: two contiguous 16B runs at p and p+16.
__device__ __forceinline__ v16bf load_a(const __bf16* p) {
  union { v16bf v; v8bf h[2]; } u;
  u.h[0] = *(const v8bf*)(p);
  u.h[1] = *(const v8bf*)(p + 16);
  return u.v;
}

__device__ __forceinline__ v8f wmma_bf16(v16bf a, v16bf b, v8f c) {
  return __builtin_amdgcn_wmma_f32_16x16x32_bf16(
      /*neg_a=*/false, a, /*neg_b=*/false, b,
      /*c_mod=*/(short)0, c, /*reuse_a=*/false, /*reuse_b=*/false);
}

// ---------------------------------------------------------------------------
// Prep kernels
// ---------------------------------------------------------------------------

// Gather item embeddings -> bf16 Epad[b][row<128][d], zero for row >= 50.
__global__ __launch_bounds__(256) void gather_kernel(
    const int* __restrict__ item_seq, const float* __restrict__ item_emb,
    __bf16* __restrict__ Epad) {
  int idx = blockIdx.x * 256 + threadIdx.x;   // CB * EROWS * 32 threads
  int d4  = idx & 31;
  int row = (idx >> 5) & (EROWS - 1);
  int b   = idx >> 12;
  if (b >= CB) return;
  v4bf o = {};
  if (row < CL) {
    int item = item_seq[b * CL + row];
    const float* src = item_emb + (size_t)item * CD + (d4 << 2);
    o[0] = (__bf16)src[0]; o[1] = (__bf16)src[1];
    o[2] = (__bf16)src[2]; o[3] = (__bf16)src[3];
  }
  *(v4bf*)(Epad + ((size_t)b * EROWS + row) * CD + (d4 << 2)) = o;
}

// Pre-swizzle Wh (L,NH,L,D) f32 -> B-fragments WhFrag[(l*50+t)*4+dc][lane][e].
__global__ __launch_bounds__(256) void whfrag_kernel(
    const float* __restrict__ Wh, __bf16* __restrict__ WhFrag) {
  int idx = blockIdx.x * 256 + threadIdx.x;   // 50*50*4*32 = 320000
  if (idx >= CL * CL * 4 * 32) return;
  int lane = idx & 31;
  int dc   = (idx >> 5) & 3;
  int rest = idx >> 7;
  int t = rest % CL;
  int l = rest / CL;
  int n = lane & 15, half = lane >> 4;
  const float* src = Wh + ((size_t)(l * CNH + n) * CL + t) * CD + dc * 32;
  v16bf o;
#pragma unroll
  for (int e = 0; e < 16; ++e) o[e] = (__bf16)src[kslot(e, half)];
  *(v16bf*)(WhFrag + (size_t)((l * CL + t) * 4 + dc) * 512 + lane * 16) = o;
}

// Pre-swizzle fc_W (1824,128) f32 -> B-fragments FcFrag[kc*8+nt][lane][e].
__global__ __launch_bounds__(256) void fcfrag_kernel(
    const float* __restrict__ fc_W, __bf16* __restrict__ FcFrag) {
  int idx = blockIdx.x * 256 + threadIdx.x;   // 57*8*32 = 14592
  if (idx >= KC_FC * 8 * 32) return;
  int lane = idx & 31;
  int nt   = (idx >> 5) & 7;
  int kc   = idx >> 8;
  int n = nt * 16 + (lane & 15), half = lane >> 4;
  v16bf o;
#pragma unroll
  for (int e = 0; e < 16; ++e) {
    int k = kc * 32 + kslot(e, half);
    o[e] = (__bf16)fc_W[(size_t)k * CD + n];
  }
  *(v16bf*)(FcFrag + (size_t)(kc * 8 + nt) * 512 + lane * 16) = o;
}

// P_u -> out[:, 128:256] (f32 passthrough).
__global__ __launch_bounds__(256) void pu_kernel(
    const int* __restrict__ user_ids, const float* __restrict__ user_emb,
    float* __restrict__ out) {
  int idx = blockIdx.x * 256 + threadIdx.x;   // CB*128
  int d = idx & 127;
  int b = idx >> 7;
  if (b >= CB) return;
  out[(size_t)b * 256 + 128 + d] = user_emb[(size_t)user_ids[b] * CD + d];
}

// o_v[b, v*128+d] = bv[v] + sum_t Wv[v,t] * item_emb[seq[b,t], d]  (f32 math)
__global__ __launch_bounds__(256) void ov_kernel(
    const int* __restrict__ item_seq, const float* __restrict__ item_emb,
    const float* __restrict__ Wv, const float* __restrict__ bv,
    __bf16* __restrict__ o_bf) {
  int idx = blockIdx.x * 256 + threadIdx.x;   // CB*NV*D = 1048576
  int d = idx & (CD - 1);
  int v = (idx >> 7) & (CNV - 1);
  int b = idx >> 10;
  if (b >= CB) return;
  const int*   seq = item_seq + b * CL;
  const float* wv  = Wv + v * CL;
  float acc = bv[v];
#pragma unroll 5
  for (int t = 0; t < CL; ++t)
    acc += wv[t] * item_emb[(size_t)seq[t] * CD + d];
  o_bf[(size_t)b * O_DIM + v * CD + d] = (__bf16)acc;
}

// ---------------------------------------------------------------------------
// Conv + ReLU + causal max. Templated on NST (number of 16-wide s-tiles) so
// the K-loop is branch-free. Block = 64 threads (2 waves, 1 batch each).
// E rows 0..65 are staged once into LDS (async-to-LDS when available); the
// 4KB of weight fragments for each t are staged per iteration and shared.
// ---------------------------------------------------------------------------
template <int NST>
__global__ __launch_bounds__(64) void conv_kernel(
    const __bf16* __restrict__ Epad, const __bf16* __restrict__ WhFrag,
    const float* __restrict__ bh, __bf16* __restrict__ o_bf, int l0) {
  __shared__ v16bf  lwb[4][32];             // weight fragments for current t
  __shared__ __bf16 sE[2][SROWS * SPITCH];  // staged E, padded pitch

  const int l    = l0 + (blockIdx.x >> 9);
  const int bg   = blockIdx.x & 511;        // batch pair
  const int tid  = threadIdx.x;
  const int wave = tid >> 5;
  const int lane = tid & 31;
  const int half = lane >> 4;
  const int mrow = lane & 15;
  const int b    = (bg << 1) + wave;

  // ---- stage E (2 batches x 66 rows x 256B) into LDS --------------------
  for (int i = tid; i < 2 * SROWS * 16; i += 64) {
    int bb  = i / (SROWS * 16);
    int rem = i % (SROWS * 16);
    int row = rem >> 4, ch = rem & 15;
    const __bf16* g =
        Epad + ((size_t)((bg << 1) + bb) * EROWS + row) * CD + (ch << 3);
    __bf16* d = &sE[bb][row * SPITCH + (ch << 3)];
#if HAVE_ASYNC
    async_copy16(g, d);
#else
    *(v8bf*)d = *(const v8bf*)g;
#endif
  }
#if HAVE_ASYNC
  __builtin_amdgcn_s_wait_asynccnt(0);
#endif
  __syncthreads();

  const __bf16* wbase = WhFrag + (size_t)l * CL * 2048;
  const __bf16* aE    = &sE[wave][mrow * SPITCH + (half << 3)];

  v8f c[NST] = {};

  for (int t = 0; t <= l; ++t) {
    __syncthreads();                        // WAR guard on lwb
    {
      const __bf16* wsrc = wbase + (size_t)t * 2048 + tid * 32;  // 64B/thread
      __bf16*       wdst = ((__bf16*)lwb) + tid * 32;
#if HAVE_ASYNC
#pragma unroll
      for (int j = 0; j < 4; ++j) async_copy16(wsrc + 8 * j, wdst + 8 * j);
      __builtin_amdgcn_s_wait_asynccnt(0);
#else
      *(v16bf*)wdst        = *(const v16bf*)wsrc;
      *(v16bf*)(wdst + 16) = *(const v16bf*)(wsrc + 16);
#endif
    }
    __syncthreads();
    if (t < l)                              // warm next t's weights in L2
      __builtin_prefetch(wbase + (size_t)(t + 1) * 2048 + tid * 32, 0, 1);

    const __bf16* arow = aE + t * SPITCH;
#pragma unroll
    for (int dc = 0; dc < 4; ++dc) {
      v16bf bf = lwb[dc][lane];
      const __bf16* ap = arow + dc * 32;
#pragma unroll
      for (int st = 0; st < NST; ++st)
        c[st] = wmma_bf16(load_a(ap + st * 16 * SPITCH), bf, c[st]);
    }
  }

  // ---- epilogue: bias + relu + causal mask + max over s ------------------
  const float bhv  = bh[l * CNH + mrow];    // n = lane&15 = column N
  const int   smax = CL - 1 - l;
  float m = -__builtin_inff();
#pragma unroll
  for (int st = 0; st < NST; ++st) {
#pragma unroll
    for (int v = 0; v < 8; ++v) {           // row M = v + 8*half
      int s = (st << 4) + (half << 3) + v;
      float h = c[st][v] + bhv;
      h = h > 0.f ? h : 0.f;
      if (s <= smax) m = fmaxf(m, h);
    }
  }
  m = fmaxf(m, __shfl_xor(m, 16, 32));      // combine lane halves (same N)
  if (half == 0)
    o_bf[(size_t)b * O_DIM + OV_DIM + l * CNH + mrow] = (__bf16)m;
}

// ---------------------------------------------------------------------------
// FC: z = relu(o @ fc_W + fc_b) -> out[:, 0:128]. One wave per (bt, nt) tile.
// ---------------------------------------------------------------------------
__global__ __launch_bounds__(32) void fc_kernel(
    const __bf16* __restrict__ o_bf, const __bf16* __restrict__ FcFrag,
    const float* __restrict__ fc_b, float* __restrict__ out) {
  int bt = blockIdx.x >> 3;                 // 64 batch tiles
  int nt = blockIdx.x & 7;                  // 8 column tiles
  int lane = threadIdx.x;
  int half = lane >> 4, mr = lane & 15;
  const __bf16* abase = o_bf + (size_t)(bt * 16 + mr) * O_DIM + (half << 3);
  v8f c = {};
  for (int kc = 0; kc < KC_FC; ++kc) {
    v16bf a  = load_a(abase + kc * 32);
    v16bf bf = *(const v16bf*)(FcFrag + (size_t)(kc * 8 + nt) * 512 + lane * 16);
    c = wmma_bf16(a, bf, c);
  }
  float bias = fc_b[nt * 16 + mr];
#pragma unroll
  for (int v = 0; v < 8; ++v) {
    int row = bt * 16 + (half << 3) + v;
    float z = c[v] + bias;
    z = z > 0.f ? z : 0.f;
    out[(size_t)row * 256 + nt * 16 + mr] = z;
  }
}

// ---------------------------------------------------------------------------
// Host launcher
// ---------------------------------------------------------------------------
extern "C" void kernel_launch(void* const* d_in, const int* in_sizes, int n_in,
                              void* d_out, int out_size, void* d_ws,
                              size_t ws_size, hipStream_t stream) {
  (void)in_sizes; (void)n_in; (void)out_size; (void)ws_size;
  const int*   user_ids = (const int*)d_in[0];
  const int*   item_seq = (const int*)d_in[1];
  const float* user_emb = (const float*)d_in[2];
  const float* item_emb = (const float*)d_in[3];
  const float* Wv       = (const float*)d_in[4];
  const float* bv       = (const float*)d_in[5];
  const float* Wh       = (const float*)d_in[6];
  const float* bh       = (const float*)d_in[7];
  const float* fc_W     = (const float*)d_in[8];
  const float* fc_b     = (const float*)d_in[9];
  float* out = (float*)d_out;

  char* ws = (char*)d_ws;
  __bf16* Epad   = (__bf16*)(ws);                 // 1024*128*128*2 = 33554432 B
  __bf16* WhFrag = (__bf16*)(ws + 33554432);      // 50*50*4*512*2  = 10240000 B
  __bf16* FcFrag = (__bf16*)(ws + 43794432);      // 57*8*512*2     =   466944 B
  __bf16* o_bf   = (__bf16*)(ws + 44261376);      // 1024*1824*2    =  3735552 B

  gather_kernel<<<(CB * EROWS * 32) / 256, 256, 0, stream>>>(item_seq, item_emb, Epad);
  whfrag_kernel<<<(CL * CL * 4 * 32 + 255) / 256, 256, 0, stream>>>(Wh, WhFrag);
  fcfrag_kernel<<<(KC_FC * 8 * 32 + 255) / 256, 256, 0, stream>>>(fc_W, FcFrag);
  pu_kernel<<<(CB * CD) / 256, 256, 0, stream>>>(user_ids, user_emb, out);
  ov_kernel<<<(CB * CNV * CD) / 256, 256, 0, stream>>>(item_seq, item_emb, Wv, bv, o_bf);

  // nst = (65 - l) >> 4 : specialize per l-range (branch-free inner loops).
  conv_kernel<4><<< 2 * 512, 64, 0, stream>>>(Epad, WhFrag, bh, o_bf, 0);   // l 0..1
  conv_kernel<3><<<16 * 512, 64, 0, stream>>>(Epad, WhFrag, bh, o_bf, 2);   // l 2..17
  conv_kernel<2><<<16 * 512, 64, 0, stream>>>(Epad, WhFrag, bh, o_bf, 18);  // l 18..33
  conv_kernel<1><<<16 * 512, 64, 0, stream>>>(Epad, WhFrag, bh, o_bf, 34);  // l 34..49

  fc_kernel<<<64 * 8, 32, 0, stream>>>(o_bf, FcFrag, fc_b, out);
}
